// MultiHeadAttention_5660766896750
// MI455X (gfx1250) — matrix operand order repaired and checked by reading, not compiled, not run on hardware
//
#include <hip/hip_runtime.h>
#include <hip/hip_bf16.h>

// MI455X (gfx1250) multi-head attention: f16 WMMA flash-attention with
// async-to-LDS double-buffered K/V tiles shared by all 8 waves of a block.
// B=2, S=4096, D=512, H=8, dh=64.

typedef __attribute__((ext_vector_type(16))) _Float16 v16h;
typedef __attribute__((ext_vector_type(8)))  float    v8f;
typedef __attribute__((ext_vector_type(8)))  unsigned v8u;
typedef __attribute__((__vector_size__(4 * sizeof(int)))) int i128v; // matches builtin param

union F16Frag { v8u u; v16h h; };

constexpr int BB = 2, S = 4096, DMODEL = 512, H = 8, DH = 64;
constexpr float LOG2E = 1.44269504088896340736f;

constexpr int TKEYS = 64;          // keys per LDS tile
constexpr int KSTR  = 72;          // K/V LDS row stride in halves (conflict-free, 16B-aligned)
constexpr int PSTR  = 66;          // per-wave P tile row stride in halves
constexpr int NT    = S / TKEYS;   // 64 key tiles
constexpr int KTILE = TKEYS * KSTR; // halves per K or V buffer

static __device__ __forceinline__ v8f wmma_f16(v16h a, v16h b, v8f c) {
    return __builtin_amdgcn_wmma_f32_16x16x32_f16(
        false, a, false, b, (short)0, c, false, false);
}

// ---- CDNA5 async global->LDS copy (16B per lane, tracked by ASYNCcnt) ----
static __device__ __forceinline__ void async_cp16(const _Float16* g, _Float16* l) {
    __builtin_amdgcn_global_load_async_to_lds_b128(
        (__attribute__((address_space(1))) i128v*)(i128v*)(void*)const_cast<_Float16*>(g),
        (__attribute__((address_space(3))) i128v*)(i128v*)(void*)l,
        /*offset=*/0, /*cpol=*/0);
}

template <int N>
static __device__ __forceinline__ void wait_async() {
#if __has_builtin(__builtin_amdgcn_s_wait_asynccnt)
    __builtin_amdgcn_s_wait_asynccnt((unsigned short)N);
#else
    asm volatile("s_wait_asynccnt %0" :: "n"(N) : "memory");
#endif
}

// ---------------- conversion kernels ----------------

__global__ void cvt_f16(const float* __restrict__ src, _Float16* __restrict__ dst, int n) {
    int i = blockIdx.x * blockDim.x + threadIdx.x;
    int stride = gridDim.x * blockDim.x;
    for (; i < n; i += stride) dst[i] = (_Float16)src[i];
}

// W: [H, 512, 64] f32 -> Wt: [H, 64, 512] f16 (transposed per head)
__global__ void cvt_w_t(const float* __restrict__ W, _Float16* __restrict__ Wt) {
    int i = blockIdx.x * blockDim.x + threadIdx.x; // exactly H*512*64 threads
    int h = i >> 15;
    int r = (i >> 6) & 511;
    int c = i & 63;
    Wt[(((size_t)h << 6) + c) * DMODEL + r] = (_Float16)W[i];
}

// ---------------- QKV projection (WMMA) ----------------
// One wave computes a 16x64 tile of Q, K or V for one (b,h).
// Q,K stored row-major [B,H,S,64]; V stored transposed [B,H,64,S].
__global__ void qkv_proj(const _Float16* __restrict__ xh,
                         const _Float16* __restrict__ Wqt,
                         const _Float16* __restrict__ Wkt,
                         const _Float16* __restrict__ Wvt,
                         const float* __restrict__ bq,
                         const float* __restrict__ bk,
                         const float* __restrict__ bv,
                         _Float16* __restrict__ Q,
                         _Float16* __restrict__ Kv,
                         _Float16* __restrict__ Vt) {
    const int lane = threadIdx.x & 31;
    const int w = (blockIdx.x * blockDim.x + threadIdx.x) >> 5; // 0..12287
    const int kind = w >> 12;           // 0=Q,1=K,2=V
    const int rem  = w & 4095;
    const int b  = rem >> 11;
    const int h  = (rem >> 8) & 7;
    const int m0 = (rem & 255) << 4;

    const _Float16* Wt   = (kind == 0) ? Wqt : (kind == 1) ? Wkt : Wvt;
    const float*    bias = (kind == 0) ? bq  : (kind == 1) ? bk  : bv;

    const int lm = lane & 15;
    const int lh = lane >> 4;

    const v8f vzero = {0.f,0.f,0.f,0.f,0.f,0.f,0.f,0.f};
    v8f acc[4];
    #pragma unroll
    for (int f = 0; f < 4; ++f) acc[f] = vzero;

    const _Float16* xrow = xh + ((size_t)b * S + (m0 + lm)) * DMODEL;
    for (int kk = 0; kk < 16; ++kk) {
        F16Frag a;
        const int klo = kk * 32 + lh * 8;
        #pragma unroll
        for (int v = 0; v < 8; ++v) {
            const int k = klo + ((v >> 2) << 4) + ((v & 3) << 1);
            a.u[v] = *(const unsigned*)(xrow + k);
        }
        const int kb = kk * 32 + lh * 8;
        #pragma unroll
        for (int f = 0; f < 4; ++f) {
            const _Float16* wrow = Wt + ((size_t)h * DH + f * 16 + lm) * DMODEL + kb;
            F16Frag bf;
            #pragma unroll
            for (int v = 0; v < 8; ++v) bf.u[v] = *(const unsigned*)(wrow + ((v >> 2) << 4) + ((v & 3) << 1));
            acc[f] = wmma_f16(a.h, bf.h, acc[f]);
        }
    }

    const size_t bh = (size_t)b * H + h;
    #pragma unroll
    for (int f = 0; f < 4; ++f) {
        const int n = f * 16 + lm;
        const float bb = bias[h * DH + n];
        #pragma unroll
        for (int v = 0; v < 8; ++v) {
            const int row = m0 + v + lh * 8;
            const _Float16 val = (_Float16)(acc[f][v] + bb);
            if (kind == 0)      Q [(bh * S + row) * DH + n] = val;
            else if (kind == 1) Kv[(bh * S + row) * DH + n] = val;
            else                Vt[(bh * DH + n) * S + row] = val;
        }
    }
}

// ---------------- flash attention (WMMA + async LDS pipeline) ----------------
// Block = 8 waves, owns (b, h, 128 query rows). K/V tiles (64 keys) are
// async-DMA'd to LDS, double buffered, shared by all waves.
// LDS layout (halves): [0, KTILE)        K buf 0
//                      [KTILE, 2K)       K buf 1
//                      [2K, 3K)          V buf 0
//                      [3K, 4K)          V buf 1
//                      [4K, ...)         8 per-wave P tiles (16 x PSTR)

static __device__ __forceinline__ void issue_tile(const _Float16* Kb, const _Float16* Vb,
                                                  _Float16* smem, int buf,
                                                  int t0, int tid) {
    _Float16* kd = smem + buf * KTILE;
    _Float16* vd = smem + 2 * KTILE + buf * KTILE;
    // K tile: 64 rows x 64 halves -> 512 x 16B chunks over 256 threads (2 instrs/wave)
    #pragma unroll
    for (int i = 0; i < 2; ++i) {
        const int c = i * 256 + tid;
        const int t = c >> 3, ko = (c & 7) << 3;
        async_cp16(Kb + (size_t)(t0 + t) * DH + ko, kd + t * KSTR + ko);
    }
    // V tile (already transposed [64, S] in global): 64 rows x 64 halves
    #pragma unroll
    for (int i = 0; i < 2; ++i) {
        const int c = i * 256 + tid;
        const int n = c >> 3, to = (c & 7) << 3;
        async_cp16(Vb + (size_t)n * S + t0 + to, vd + n * KSTR + to);
    }
}

__global__ void __launch_bounds__(256)
flash_attn(const _Float16* __restrict__ Q,
           const _Float16* __restrict__ Kv,
           const _Float16* __restrict__ Vt,
           float* __restrict__ out) {
    __shared__ __attribute__((aligned(16))) _Float16 smem[4 * KTILE + 8 * 16 * PSTR];

    const int tid  = threadIdx.x;
    const int lane = tid & 31;
    const int wib  = tid >> 5;
    const int bb   = blockIdx.x;                  // 0..511
    const int b  = bb >> 8;
    const int h  = (bb >> 5) & 7;
    const int m0 = ((bb & 31) << 7) + (wib << 4); // this wave's 16 query rows

    const int lm = lane & 15;
    const int lh = lane >> 4;
    const size_t bh = (size_t)b * H + h;
    const _Float16* Qb = Q  + bh * (size_t)S * DH;
    const _Float16* Kb = Kv + bh * (size_t)S * DH;
    const _Float16* Vb = Vt + bh * (size_t)DH * S;

    // Q A-fragments (dk=64 -> 2 chunks of K=32), pre-scaled by 1/sqrt(64) (exact in f16)
    F16Frag qf[2];
    {
        const _Float16* qrow = Qb + (size_t)(m0 + lm) * DH;
        #pragma unroll
        for (int kk = 0; kk < 2; ++kk) {
            #pragma unroll
            for (int v = 0; v < 8; ++v) {
                const int k = kk * 32 + lh * 8 + ((v >> 2) << 4) + ((v & 3) << 1);
                qf[kk].u[v] = *(const unsigned*)(qrow + k);
            }
            #pragma unroll
            for (int i = 0; i < 16; ++i) qf[kk].h[i] *= (_Float16)0.125f;
        }
    }

    const v8f vzero = {0.f,0.f,0.f,0.f,0.f,0.f,0.f,0.f};
    v8f acc[4];
    #pragma unroll
    for (int f = 0; f < 4; ++f) acc[f] = vzero;
    float mrow[8], lrow[8];
    #pragma unroll
    for (int v = 0; v < 8; ++v) { mrow[v] = -3.0e38f; lrow[v] = 0.0f; }

    _Float16* pt = smem + 4 * KTILE + wib * (16 * PSTR);

    // Prologue: prefetch tiles 0 and 1
    issue_tile(Kb, Vb, smem, 0, 0,     tid);
    issue_tile(Kb, Vb, smem, 1, TKEYS, tid);

    for (int tb = 0; tb < NT; ++tb) {
        // Retire this tile's 4 async ops (next tile's 4 may stay in flight)
        if (tb + 1 < NT) wait_async<4>(); else wait_async<0>();
        __syncthreads();

        const int buf = tb & 1;
        _Float16* kb = smem + buf * KTILE;
        _Float16* vb = smem + 2 * KTILE + buf * KTILE;

        // S = Q * K^T for 64 keys (four 16-key D fragments), B-frags from LDS
        v8f sf[4];
        #pragma unroll
        for (int j = 0; j < 4; ++j) {
            sf[j] = vzero;
            const _Float16* krow = kb + (j * 16 + lm) * KSTR + lh * 8;
            #pragma unroll
            for (int kk = 0; kk < 2; ++kk) {
                F16Frag bf;
                #pragma unroll
                for (int v = 0; v < 8; ++v)
                    bf.u[v] = *(const unsigned*)(krow + kk * 32 + ((v >> 2) << 4) + ((v & 3) << 1));
                sf[j] = wmma_f16(qf[kk].h, bf.h, sf[j]);
            }
        }

        // Online softmax over the 64-key block (rows live per-VGPR in half-waves)
        #pragma unroll
        for (int v = 0; v < 8; ++v) {
            float mb = fmaxf(fmaxf(sf[0][v], sf[1][v]), fmaxf(sf[2][v], sf[3][v]));
            #pragma unroll
            for (int off = 8; off > 0; off >>= 1)
                mb = fmaxf(mb, __shfl_xor(mb, off, 32));
            const float mn = fmaxf(mrow[v], mb);
            const float al = __builtin_amdgcn_exp2f((mrow[v] - mn) * LOG2E);
            float rs = 0.0f;
            #pragma unroll
            for (int j = 0; j < 4; ++j) {
                const float p = __builtin_amdgcn_exp2f((sf[j][v] - mn) * LOG2E);
                sf[j][v] = p;
                rs += p;
            }
            #pragma unroll
            for (int off = 8; off > 0; off >>= 1)
                rs += __shfl_xor(rs, off, 32);
            lrow[v] = lrow[v] * al + rs;
            mrow[v] = mn;
            #pragma unroll
            for (int f = 0; f < 4; ++f) acc[f][v] *= al;
        }

        // Spill P (16x64 f16) to this wave's LDS tile (D-frag layout -> row major)
        #pragma unroll
        for (int j = 0; j < 4; ++j) {
            #pragma unroll
            for (int v = 0; v < 8; ++v)
                pt[(v + lh * 8) * PSTR + j * 16 + lm] = (_Float16)sf[j][v];
        }

        // acc += P(16x64) * V(64x64): two K=32 A-chunks x four dv blocks
        #pragma unroll
        for (int g = 0; g < 2; ++g) {
            F16Frag pf;
            const _Float16* prow = pt + lm * PSTR + g * 32 + lh * 8;
            #pragma unroll
            for (int v = 0; v < 8; ++v)
                pf.u[v] = *(const unsigned*)(prow + ((v >> 2) << 4) + ((v & 3) << 1));
            #pragma unroll
            for (int f = 0; f < 4; ++f) {
                const _Float16* vrow = vb + (f * 16 + lm) * KSTR + g * 32 + lh * 8;
                F16Frag bf;
                #pragma unroll
                for (int v = 0; v < 8; ++v) bf.u[v] = *(const unsigned*)(vrow + ((v >> 2) << 4) + ((v & 3) << 1));
                acc[f] = wmma_f16(pf.h, bf.h, acc[f]);
            }
        }

        __syncthreads(); // everyone done reading buf before it is refilled
        if (tb + 2 < NT)
            issue_tile(Kb, Vb, smem, buf, (tb + 2) * TKEYS, tid);
    }

    // Epilogue: normalize and write f32 output [B, S, H*64]
    #pragma unroll
    for (int f = 0; f < 4; ++f) {
        const int n = f * 16 + lm;
        #pragma unroll
        for (int v = 0; v < 8; ++v) {
            const int row = m0 + v + lh * 8;
            out[((size_t)b * S + row) * (H * DH) + h * DH + n] = acc[f][v] / lrow[v];
        }
    }
}

// ---------------- host launch ----------------

extern "C" void kernel_launch(void* const* d_in, const int* in_sizes, int n_in,
                              void* d_out, int out_size, void* d_ws, size_t ws_size,
                              hipStream_t stream) {
    const float* x  = (const float*)d_in[0];
    const float* Wq = (const float*)d_in[1];
    const float* Wk = (const float*)d_in[2];
    const float* Wv = (const float*)d_in[3];
    const float* bq = (const float*)d_in[4];
    const float* bk = (const float*)d_in[5];
    const float* bv = (const float*)d_in[6];
    float* out = (float*)d_out;

    char* ws = (char*)d_ws;
    _Float16* xh  = (_Float16*)ws; ws += (size_t)BB * S * DMODEL * sizeof(_Float16);
    _Float16* Wqt = (_Float16*)ws; ws += (size_t)H * DH * DMODEL * sizeof(_Float16);
    _Float16* Wkt = (_Float16*)ws; ws += (size_t)H * DH * DMODEL * sizeof(_Float16);
    _Float16* Wvt = (_Float16*)ws; ws += (size_t)H * DH * DMODEL * sizeof(_Float16);
    _Float16* Qh  = (_Float16*)ws; ws += (size_t)BB * H * S * DH * sizeof(_Float16);
    _Float16* Kh  = (_Float16*)ws; ws += (size_t)BB * H * S * DH * sizeof(_Float16);
    _Float16* Vth = (_Float16*)ws; ws += (size_t)BB * H * S * DH * sizeof(_Float16);

    cvt_f16<<<2048, 256, 0, stream>>>(x, xh, BB * S * DMODEL);
    cvt_w_t<<<1024, 256, 0, stream>>>(Wq, Wqt);   // H*512*64 = 262144 threads exactly
    cvt_w_t<<<1024, 256, 0, stream>>>(Wk, Wkt);
    cvt_w_t<<<1024, 256, 0, stream>>>(Wv, Wvt);

    // 3 * B*H*(S/16) = 12288 waves
    qkv_proj<<<1536, 256, 0, stream>>>(xh, Wqt, Wkt, Wvt, bq, bk, bv, Qh, Kh, Vth);

    // B*H*(S/128) = 512 blocks x 8 waves
    flash_attn<<<512, 256, 0, stream>>>(Qh, Kh, Vth, out);
}
